// multiAttentionHead_49701361549904
// MI455X (gfx1250) — compile-verified, matrix operand order checked by reading, not run on hardware
//
#include <hip/hip_runtime.h>

// Problem constants (match reference)
#define B_  2
#define S_  4096
#define F_  640
#define H_  10
#define D_  64
#define HD_ 640          // H_*D_
#define M_  (B_*S_)      // 8192

#define AS1 __attribute__((address_space(1)))
#define AS3 __attribute__((address_space(3)))

typedef __attribute__((ext_vector_type(16))) _Float16 v16h;
typedef __attribute__((ext_vector_type(8)))  _Float16 v8h;
typedef __attribute__((ext_vector_type(8)))  float    v8f;
typedef __attribute__((ext_vector_type(4)))  unsigned int v4u;
typedef __attribute__((ext_vector_type(8)))  int      v8i_;
typedef __attribute__((ext_vector_type(4)))  int      v4i_;

#if __has_builtin(__builtin_amdgcn_tensor_load_to_lds) && \
    __has_builtin(__builtin_amdgcn_s_wait_tensorcnt)
#define USE_TDM 1
#else
#define USE_TDM 0
#endif

#if __has_builtin(__builtin_amdgcn_global_load_async_to_lds_b128) && \
    __has_builtin(__builtin_amdgcn_s_wait_asynccnt)
#define USE_ASYNC_LDS 1
#else
#define USE_ASYNC_LDS 0
#endif

// LDS row strides (in halfs) padded to break bank conflicts while keeping
// every 8-half fragment chunk 16B-aligned.
#define LDSAB_STR 40   // GEMM A/B tiles: 32 data + 8 pad  (80B rows)
#define LDSK_STR  80   // attn K tile:    64 data + 16 pad (160B rows)
#define LDSV_STR  40   // attn V tile:    32 data + 8 pad  (80B rows)
#define LDSP_STR  40   // attn P tile:    32 data + 8 pad  (80B rows)

// Load one wave-lane's slice of a 16x32 f16 WMMA A/B fragment.
// CDNA5 layout: lanes 0-15 hold K[0..7] (v0..3) and K[16..23] (v4..7),
// lanes 16-31 hold K[8..15] and K[24..31]  ->  two contiguous 8-half chunks
// at rowbase+c0 and rowbase+16+c0 with c0 = (lane>>4)*8.
__device__ __forceinline__ v16h load_frag(const _Float16* rowbase, int c0) {
  v8h lo = *(const v8h*)(rowbase + c0);
  v8h hi = *(const v8h*)(rowbase + 16 + c0);
  return __builtin_shufflevector(lo, hi,
      0,1,2,3,4,5,6,7,8,9,10,11,12,13,14,15);
}

#if USE_TDM
// Build a D# for a 2D f16 tile (tile_d1 rows of tile_d0 elements, row stride
// stride0 elements) copied from global memory to LDS, inserting LDS padding
// (pad_interval/pad_amount codes per ISA 08 §8.4), and issue
// TENSOR_LOAD_TO_LDS. All arguments must be workgroup-uniform.
__device__ __forceinline__ void tdm_load_2d_f16(const void* lds_dst,
                                                const _Float16* gsrc,
                                                unsigned tile_d0,
                                                unsigned tile_d1,
                                                unsigned long long stride0,
                                                unsigned pad_interval_code,
                                                unsigned pad_amount_code) {
  unsigned lds_off = (unsigned)(unsigned long long)(const AS3 void*)lds_dst;
  unsigned long long ga = (unsigned long long)gsrc;
  v4u g0;
  g0.x = 1u;                                           // count=1, user mode
  g0.y = lds_off;                                      // lds_addr (bytes)
  g0.z = (unsigned)(ga & 0xffffffffu);                 // global_addr[31:0]
  g0.w = (unsigned)((ga >> 32) & 0x1ffffffu) | (2u << 30); // [56:32] + type=2
  unsigned td0 = tile_d0, td1 = tile_d1;
  v8i_ g1;
  g1[0] = (int)((1u << 16) |                           // data_size=1 (2 bytes)
                (1u << 20) |                           // pad_enable
                (pad_interval_code << 22) |
                (pad_amount_code << 25));
  g1[1] = (int)((td0 & 0xffffu) << 16);                // tensor_dim0[15:0]
  g1[2] = (int)(((td0 >> 16) & 0xffffu) | ((td1 & 0xffffu) << 16));
  g1[3] = (int)(((td1 >> 16) & 0xffffu) | ((tile_d0 & 0xffffu) << 16));
  g1[4] = (int)(tile_d1 & 0xffffu);                    // tile_dim1, tile_dim2=0
  g1[5] = (int)(unsigned)(stride0 & 0xffffffffu);      // dim0_stride[31:0]
  g1[6] = (int)(unsigned)((stride0 >> 32) & 0xffffu);  // dim0_stride[47:32]
  g1[7] = 0;
  v4i_ z4 = {0, 0, 0, 0};
  v8i_ z8 = {0, 0, 0, 0, 0, 0, 0, 0};
  __builtin_amdgcn_tensor_load_to_lds(g0, g1, z4, z4, z8, 0);
}
#endif

__global__ void cvt_f32_f16(const float* __restrict__ in,
                            _Float16* __restrict__ out, int n) {
  int i = blockIdx.x * blockDim.x + threadIdx.x;
  int stride = gridDim.x * blockDim.x;
  for (; i < n; i += stride) out[i] = (_Float16)in[i];
}

// out[c][r] = (f16) in[r][c]  -- LDS-tiled transpose, coalesced both ways.
// Grid: (C/32, R/32), block (32, 8).
__global__ __launch_bounds__(256)
void transpose_cvt(const float* __restrict__ in, _Float16* __restrict__ out,
                   int R, int C) {
  __shared__ float tile[32][33];
  int r0 = blockIdx.y * 32, c0 = blockIdx.x * 32;
  #pragma unroll
  for (int i = threadIdx.y; i < 32; i += 8)
    tile[i][threadIdx.x] = in[(size_t)(r0 + i) * C + c0 + threadIdx.x];
  __syncthreads();
  #pragma unroll
  for (int i = threadIdx.y; i < 32; i += 8)
    out[(size_t)(c0 + i) * R + r0 + threadIdx.x] = (_Float16)tile[threadIdx.x][i];
}

// C[M,N] = A[M,K] * Bt[N,K]^T, f16 inputs (B pre-transposed), f32 accumulate.
// 2-deep ping-pong pipelined: tile k+1 streams into LDS (async DMA) behind
// the WMMAs on tile k; s_wait_asynccnt(4) = threshold wait (in-order compl).
// mode 0: store f16 to [B,H,S,D]   (Q, K)
// mode 1: store f16 to [B,H,D,S]   (V transposed)
// mode 2: store f32 row-major [M,N] (final output)
__global__ __launch_bounds__(128)
void gemm_f16(const _Float16* __restrict__ A, const _Float16* __restrict__ Bt,
              void* __restrict__ Cout, int Mdim, int Ndim, int Kdim, int mode) {
  __shared__ __align__(16) _Float16 ldsA[2][64 * LDSAB_STR];
  __shared__ __align__(16) _Float16 ldsB[2][64 * LDSAB_STR];  // [n][k]
  const int tid  = threadIdx.x;
  const int wid  = tid >> 5;
  const int lane = tid & 31;
  const int r16  = lane & 15, hi = lane >> 4;
  const int m0 = blockIdx.y * 64;
  const int n0 = blockIdx.x * 64;
  const int wm = (wid & 1) * 32;
  const int wn = (wid >> 1) * 32;

  // Stage one 64x32 A tile + 64x32 B tile into buffer `buf`.
  auto stage = [&](int k0, int buf) {
#if USE_ASYNC_LDS
    #pragma unroll
    for (int idx = tid; idx < 256; idx += 128) {
      int row = idx >> 2, seg = idx & 3;
      __builtin_amdgcn_global_load_async_to_lds_b128(
          (AS1 v4i_*)(A + (size_t)(m0 + row) * Kdim + k0 + seg * 8),
          (AS3 v4i_*)&ldsA[buf][row * LDSAB_STR + seg * 8], 0, 0);
      __builtin_amdgcn_global_load_async_to_lds_b128(
          (AS1 v4i_*)(Bt + (size_t)(n0 + row) * Kdim + k0 + seg * 8),
          (AS3 v4i_*)&ldsB[buf][row * LDSAB_STR + seg * 8], 0, 0);
    }
#else
    {
      int row0 = tid >> 2,         seg0 = tid & 3;
      int row1 = (tid + 128) >> 2, seg1 = (tid + 128) & 3;
      v8h a0 = *(const v8h*)(A  + (size_t)(m0 + row0) * Kdim + k0 + seg0 * 8);
      v8h a1 = *(const v8h*)(A  + (size_t)(m0 + row1) * Kdim + k0 + seg1 * 8);
      v8h b0 = *(const v8h*)(Bt + (size_t)(n0 + row0) * Kdim + k0 + seg0 * 8);
      v8h b1 = *(const v8h*)(Bt + (size_t)(n0 + row1) * Kdim + k0 + seg1 * 8);
      *(v8h*)&ldsA[buf][row0 * LDSAB_STR + seg0 * 8] = a0;
      *(v8h*)&ldsA[buf][row1 * LDSAB_STR + seg1 * 8] = a1;
      *(v8h*)&ldsB[buf][row0 * LDSAB_STR + seg0 * 8] = b0;
      *(v8h*)&ldsB[buf][row1 * LDSAB_STR + seg1 * 8] = b1;
    }
#endif
  };

  v8f acc[2][2] = {};
  const int nk = Kdim / 32;

  stage(0, 0);  // prologue: tile 0 -> buf 0

  for (int kt = 0; kt < nk; ++kt) {
    const int cur = kt & 1;
    if (kt + 1 < nk) {
      stage((kt + 1) * 32, cur ^ 1);      // next tile streams behind compute
#if USE_ASYNC_LDS
      __builtin_amdgcn_s_wait_asynccnt(4);  // tile kt resident; kt+1 in flight
#endif
    } else {
#if USE_ASYNC_LDS
      __builtin_amdgcn_s_wait_asynccnt(0);
#endif
    }
    __syncthreads();  // publish buf[cur]

    v16h af[2], bf[2];
    #pragma unroll
    for (int sm = 0; sm < 2; ++sm)
      af[sm] = load_frag(&ldsA[cur][(wm + sm * 16 + r16) * LDSAB_STR], hi * 8);
    #pragma unroll
    for (int sn = 0; sn < 2; ++sn)
      bf[sn] = load_frag(&ldsB[cur][(wn + sn * 16 + r16) * LDSAB_STR], hi * 8);
    #pragma unroll
    for (int sm = 0; sm < 2; ++sm)
      #pragma unroll
      for (int sn = 0; sn < 2; ++sn)
        acc[sm][sn] = __builtin_amdgcn_wmma_f32_16x16x32_f16(
            false, af[sm], false, bf[sn], (short)0, acc[sm][sn], false, false);
    __syncthreads();  // reads of buf[cur] done (guards overwrite at kt+2)
  }

  // D fragment: lane holds col n = r16, rows m = r + 8*hi (r = 0..7)
  #pragma unroll
  for (int sm = 0; sm < 2; ++sm)
    #pragma unroll
    for (int sn = 0; sn < 2; ++sn)
      #pragma unroll
      for (int r = 0; r < 8; ++r) {
        int m = m0 + wm + sm * 16 + r + 8 * hi;
        int n = n0 + wn + sn * 16 + r16;
        float v = acc[sm][sn][r];
        if (mode == 2) {
          ((float*)Cout)[(size_t)m * Ndim + n] = v;
        } else {
          int b = m / S_, s = m % S_;
          int h = n / D_, d = n % D_;
          size_t off = (mode == 0)
              ? ((size_t)((b * H_ + h) * S_) + s) * D_ + d
              : ((size_t)((b * H_ + h) * D_) + d) * S_ + s;
          ((_Float16*)Cout)[off] = (_Float16)v;
        }
      }
}

// Fused flash attention. Q,K: [B,H,S,D] f16; Vt: [B,H,D,S] f16.
// Ctx out: [B,S,H,D] f16. 256 threads = 8 waves, each wave owns 16 q rows.
// K/V chunks are TDM ping-pong pipelined: wave 0 issues the next chunk's two
// descriptors, then s_wait_tensorcnt(2) guarantees the current chunk is
// resident while the next streams in behind the WMMAs + softmax.
__global__ __launch_bounds__(256)
void attn_fused(const _Float16* __restrict__ Q, const _Float16* __restrict__ K,
                const _Float16* __restrict__ Vt, _Float16* __restrict__ Ctx) {
  __shared__ __align__(16) _Float16 ldsK[2][32 * LDSK_STR];   // [k][d] padded
  __shared__ __align__(16) _Float16 ldsV[2][64 * LDSV_STR];   // [d][k] padded
  __shared__ __align__(16) _Float16 ldsP[8 * 16 * LDSP_STR];  // per-wave P
  const int tid  = threadIdx.x;
  const int wid  = tid >> 5;
  const int lane = tid & 31;
  const int r16  = lane & 15, hi = lane >> 4;
  const int h = blockIdx.y, b = blockIdx.z;
  const int qbase = blockIdx.x * 128 + wid * 16;
  const size_t bh = (size_t)(b * H_ + h);
  const _Float16* Qb = Q  + bh * S_ * D_;
  const _Float16* Kb = K  + bh * S_ * D_;
  const _Float16* Vb = Vt + bh * D_ * S_;

  // Stage K chunk (32x64) + Vt chunk (64x32) for k-base `kc` into buffer.
  auto stage = [&](int kc, int buf) {
#if USE_TDM
    if (wid == 0) {
      // K chunk: contiguous -> 1D copy of 2048 elems; pad 32B every 128B
      tdm_load_2d_f16(&ldsK[buf][0], Kb + (size_t)kc * 64, 2048u, 1u, 2048ull,
                      4u /*128B*/, 7u /*8 dwords*/);
      // Vt chunk: 64 rows x 32 halfs, row stride S_; pad 16B every 64B
      tdm_load_2d_f16(&ldsV[buf][0], Vb + kc, 32u, 64u, (unsigned long long)S_,
                      3u /*64B*/, 3u /*4 dwords*/);
    }
#else
    {
      int kk = tid >> 3, segk = tid & 7;
      int d  = tid >> 2, segv = tid & 3;
      v8h tk = *(const v8h*)(Kb + (size_t)(kc + kk) * 64 + segk * 8);
      v8h tv = *(const v8h*)(Vb + (size_t)d * S_ + kc + segv * 8);
      *(v8h*)&ldsK[buf][kk * LDSK_STR + segk * 8] = tk;
      *(v8h*)&ldsV[buf][d * LDSV_STR + segv * 8] = tv;
    }
#endif
  };

  // Q fragments for d=0..31 and d=32..63, pre-scaled by 1/sqrt(D)=0.125
  const _Float16* qrow = Qb + (size_t)(qbase + r16) * D_;
  v16h aq0 = load_frag(qrow, hi * 8);
  v16h aq1 = load_frag(qrow + 32, hi * 8);
  #pragma unroll
  for (int i = 0; i < 16; ++i) {
    aq0[i] *= (_Float16)0.125f;
    aq1[i] *= (_Float16)0.125f;
  }

  v8f acc[4] = {};
  float mrow[8], lrow[8];
  #pragma unroll
  for (int r = 0; r < 8; ++r) { mrow[r] = -3.0e38f; lrow[r] = 0.0f; }

  _Float16* Pw = &ldsP[wid * 16 * LDSP_STR];
  const int nit = S_ / 32;

  stage(0, 0);  // prologue: chunk 0 -> buf 0

  for (int it = 0; it < nit; ++it) {
    const int cur = it & 1;
    if (it + 1 < nit) {
      stage((it + 1) * 32, cur ^ 1);      // next chunk streams behind compute
#if USE_TDM
      if (wid == 0) __builtin_amdgcn_s_wait_tensorcnt(2);
#endif
    } else {
#if USE_TDM
      if (wid == 0) __builtin_amdgcn_s_wait_tensorcnt(0);
#endif
    }
    __syncthreads();  // buf[cur] resident & published
    const _Float16* Kc = &ldsK[cur][0];
    const _Float16* Vc = &ldsV[cur][0];

    // scores: S[16q x 32k] = Q[16x64] * K^T[64x32]  (4 WMMAs)
    v8f s0 = {}, s1 = {};
    s0 = __builtin_amdgcn_wmma_f32_16x16x32_f16(false, aq0, false,
             load_frag(&Kc[r16 * LDSK_STR], hi * 8), (short)0, s0, false, false);
    s0 = __builtin_amdgcn_wmma_f32_16x16x32_f16(false, aq1, false,
             load_frag(&Kc[r16 * LDSK_STR + 32], hi * 8), (short)0, s0, false, false);
    s1 = __builtin_amdgcn_wmma_f32_16x16x32_f16(false, aq0, false,
             load_frag(&Kc[(16 + r16) * LDSK_STR], hi * 8), (short)0, s1, false, false);
    s1 = __builtin_amdgcn_wmma_f32_16x16x32_f16(false, aq1, false,
             load_frag(&Kc[(16 + r16) * LDSK_STR + 32], hi * 8), (short)0, s1, false, false);

    // online softmax: row m lives in vgpr r across a 16-lane half
    float p0[8], p1[8];
    #pragma unroll
    for (int r = 0; r < 8; ++r) {
      float cm = fmaxf(s0[r], s1[r]);
      #pragma unroll
      for (int msk = 1; msk <= 8; msk <<= 1)
        cm = fmaxf(cm, __shfl_xor(cm, msk, 32));
      float mn    = fmaxf(mrow[r], cm);
      float alpha = __expf(mrow[r] - mn);
      mrow[r] = mn;
      p0[r] = __expf(s0[r] - mn);
      p1[r] = __expf(s1[r] - mn);
      lrow[r] = lrow[r] * alpha + p0[r] + p1[r];  // per-lane partial row sum
      #pragma unroll
      for (int g = 0; g < 4; ++g) acc[g][r] *= alpha;
    }

    // transpose P (D-frag layout) into A-frag layout via per-wave LDS
    #pragma unroll
    for (int r = 0; r < 8; ++r) {
      int m = r + 8 * hi;
      Pw[m * LDSP_STR + r16]      = (_Float16)p0[r];
      Pw[m * LDSP_STR + 16 + r16] = (_Float16)p1[r];
    }
    __syncthreads();  // P published

    // ctx += P[16x32] * V[32k x 64d]  (4 WMMAs)
    v16h ap = load_frag(&Pw[r16 * LDSP_STR], hi * 8);
    #pragma unroll
    for (int g = 0; g < 4; ++g) {
      v16h bv = load_frag(&Vc[(g * 16 + r16) * LDSV_STR], hi * 8);
      acc[g] = __builtin_amdgcn_wmma_f32_16x16x32_f16(
          false, ap, false, bv, (short)0, acc[g], false, false);
    }
    __syncthreads();  // reads of buf[cur] done (guards overwrite at it+2)
  }

  // finalize: complete row sums across the 16-lane half, normalize, store
  #pragma unroll
  for (int r = 0; r < 8; ++r) {
    #pragma unroll
    for (int msk = 1; msk <= 8; msk <<= 1)
      lrow[r] += __shfl_xor(lrow[r], msk, 32);
    lrow[r] = 1.0f / lrow[r];
  }
  #pragma unroll
  for (int g = 0; g < 4; ++g)
    #pragma unroll
    for (int r = 0; r < 8; ++r) {
      int q = qbase + r + 8 * hi;
      int d = g * 16 + r16;
      Ctx[((size_t)(b * S_ + q) * H_ + h) * D_ + d] =
          (_Float16)(acc[g][r] * lrow[r]);
    }
}

extern "C" void kernel_launch(void* const* d_in, const int* in_sizes, int n_in,
                              void* d_out, int out_size, void* d_ws, size_t ws_size,
                              hipStream_t stream) {
  (void)in_sizes; (void)n_in; (void)out_size; (void)ws_size;
  const float* x  = (const float*)d_in[0];
  const float* wq = (const float*)d_in[1];
  const float* wk = (const float*)d_in[2];
  const float* wv = (const float*)d_in[3];
  const float* wo = (const float*)d_in[4];

  // Carve f16 staging buffers out of workspace (256B aligned, ~56 MB total)
  char* p = (char*)d_ws;
  auto carve = [&](size_t nelem) {
    _Float16* r = (_Float16*)p;
    p += (nelem * sizeof(_Float16) + 255) & ~(size_t)255;
    return r;
  };
  _Float16* x16   = carve((size_t)B_ * S_ * F_);
  _Float16* wqt16 = carve((size_t)HD_ * F_);   // Wq^T [HD][F]
  _Float16* wkt16 = carve((size_t)HD_ * F_);   // Wk^T
  _Float16* wvt16 = carve((size_t)HD_ * F_);   // Wv^T
  _Float16* wot16 = carve((size_t)F_ * HD_);   // Wo^T [F][HD]
  _Float16* q16   = carve((size_t)B_ * H_ * S_ * D_);
  _Float16* k16   = carve((size_t)B_ * H_ * S_ * D_);
  _Float16* vt16  = carve((size_t)B_ * H_ * D_ * S_);
  _Float16* c16   = carve((size_t)B_ * S_ * H_ * D_);

  cvt_f32_f16<<<2048, 256, 0, stream>>>(x, x16, B_ * S_ * F_);

  dim3 tb(32, 8);
  dim3 tgq(HD_ / 32, F_ / 32);   // Wq/Wk/Wv: [F][HD] -> [HD][F]
  transpose_cvt<<<tgq, tb, 0, stream>>>(wq, wqt16, F_, HD_);
  transpose_cvt<<<tgq, tb, 0, stream>>>(wk, wkt16, F_, HD_);
  transpose_cvt<<<tgq, tb, 0, stream>>>(wv, wvt16, F_, HD_);
  dim3 tgo(F_ / 32, HD_ / 32);   // Wo: [HD][F] -> [F][HD]
  transpose_cvt<<<tgo, tb, 0, stream>>>(wo, wot16, HD_, F_);

  dim3 gq(HD_ / 64, M_ / 64);  // (10, 128)
  gemm_f16<<<gq, 128, 0, stream>>>(x16, wqt16, q16,  M_, HD_, F_, 0);
  gemm_f16<<<gq, 128, 0, stream>>>(x16, wkt16, k16,  M_, HD_, F_, 0);
  gemm_f16<<<gq, 128, 0, stream>>>(x16, wvt16, vt16, M_, HD_, F_, 1);

  dim3 ga(S_ / 128, H_, B_);   // (32, 10, 2)
  attn_fused<<<ga, 256, 0, stream>>>(q16, k16, vt16, c16);

  dim3 go(F_ / 64, M_ / 64);   // (10, 128)
  gemm_f16<<<go, 128, 0, stream>>>(c16, wot16, d_out, M_, F_, HD_, 2);
}
